// LuluAttention_77481210020265
// MI455X (gfx1250) — compile-verified, matrix operand order checked by reading, not run on hardware
//
#include <hip/hip_runtime.h>
#include <hip/hip_bf16.h>

// ---------------------------------------------------------------- constants
constexpr int BB   = 2;
constexpr int SS   = 2048;
constexpr int HID  = 2048;
constexpr int NH   = 16;
constexpr int NKV  = 4;
constexpr int HD   = 128;
constexpr int MTOK = BB * SS;     // 4096 tokens
constexpr int KVD  = NKV * HD;    // 512

typedef __attribute__((ext_vector_type(16))) _Float16 v16h;
typedef __attribute__((ext_vector_type(8)))  float    v8f;
typedef __attribute__((ext_vector_type(4)))  unsigned int u32x4;
typedef __attribute__((ext_vector_type(4)))  int      v4i;

union Frag16 { v16h h; u32x4 q[2]; };

__device__ __forceinline__ v8f wmma_f16(const Frag16& a, const Frag16& b, v8f c) {
  return __builtin_amdgcn_wmma_f32_16x16x32_f16(false, a.h, false, b.h,
                                                (short)0, c, false, false);
}

__device__ __forceinline__ float halfmax(float v) {
#pragma unroll
  for (int off = 1; off < 16; off <<= 1) v = fmaxf(v, __shfl_xor(v, off, 32));
  return v;
}
__device__ __forceinline__ float halfsum(float v) {
#pragma unroll
  for (int off = 1; off < 16; off <<= 1) v += __shfl_xor(v, off, 32);
  return v;
}

// ---------------------------------------------- async global->LDS (guarded)
#if defined(__has_builtin)
#if __has_builtin(__builtin_amdgcn_global_load_async_to_lds_b128) && \
    __has_builtin(__builtin_amdgcn_s_wait_asynccnt)
#define USE_ASYNC 1
#endif
#endif
#ifndef USE_ASYNC
#define USE_ASYNC 0
#endif

#if USE_ASYNC
// signature (from hipcc diagnostic): (int4 AS1*, int4 AS3*, imm, imm)
typedef __attribute__((address_space(1))) v4i* gv4i_p;
typedef __attribute__((address_space(3))) v4i* lv4i_p;
__device__ __forceinline__ void async_copy16(const _Float16* g, _Float16* l) {
  __builtin_amdgcn_global_load_async_to_lds_b128((gv4i_p)g, (lv4i_p)l, 0, 0);
}
#define ASYNC_WAIT(n) __builtin_amdgcn_s_wait_asynccnt(n)
#else
__device__ __forceinline__ void async_copy16(const _Float16* g, _Float16* l) {
  *(u32x4*)l = *(const u32x4*)g;   // global_load_b128 + ds_store_b128 fallback
}
#define ASYNC_WAIT(n) ((void)0)
#endif

// ------------------------------------------------------- elementwise helpers
__global__ void k_cvt(const float* __restrict__ in, _Float16* __restrict__ out, int n) {
  int i = blockIdx.x * blockDim.x + threadIdx.x;
  if (i < n) out[i] = (_Float16)in[i];
}

// W[K,N] f32 -> Wt[N,K] f16
__global__ void k_tcvt(const float* __restrict__ W, _Float16* __restrict__ Wt,
                       int K, int N) {
  int idx = blockIdx.x * blockDim.x + threadIdx.x;
  if (idx >= K * N) return;
  int k = idx / N, n = idx % N;
  Wt[(size_t)n * K + k] = (_Float16)W[idx];
}

// ----------------------------------------------------------------- WMMA GEMM
// C[M,N] = A[M,K] (f16, row-major) x Bt[N,K] (f16, B pre-transposed)
// block = 256 threads (8 waves); block tile 128x128; k-step 64; LDS staged,
// double buffered, async global->LDS. Wave tile 32x64; 16 WMMA per barrier.
template <bool F32OUT>
__global__ __launch_bounds__(256, 1)
void k_gemm(const _Float16* __restrict__ A, const _Float16* __restrict__ Bt,
            void* __restrict__ C, int Ndim, int Kdim) {
  __shared__ _Float16 As[2][128 * 64];   // 16KB x2
  __shared__ _Float16 Bs[2][128 * 64];   // 16KB x2
  const int tid  = threadIdx.x;
  const int lane = tid & 31;
  const int wave = tid >> 5;
  const int wm   = wave & 3;       // 4 waves along M (32 rows each)
  const int wn   = wave >> 2;      // 2 waves along N (64 cols each)
  const int row  = lane & 15;
  const int koff = (lane >> 4) * 8;
  const int m0   = blockIdx.x * 128;
  const int n0   = blockIdx.y * 128;

  // staging: 1024 16B-chunks per 128x64 tile, 4 A-chunks + 4 B-chunks/thread
  auto stage = [&](int k, int buf) {
#pragma unroll
    for (int j = 0; j < 4; ++j) {
      int c = tid + j * 256;
      int r = c >> 3, o = (c & 7) * 8;
      async_copy16(A  + (size_t)(m0 + r) * Kdim + k + o, &As[buf][r * 64 + o]);
      async_copy16(Bt + (size_t)(n0 + r) * Kdim + k + o, &Bs[buf][r * 64 + o]);
    }
  };

  v8f acc[2][4] = {};
  stage(0, 0);
  const int nsteps = Kdim / 64;
  for (int i = 0; i < nsteps; ++i) {
    const int buf = i & 1;
    if (i + 1 < nsteps) { stage((i + 1) * 64, buf ^ 1); ASYNC_WAIT(8); }
    else                { ASYNC_WAIT(0); }
    __syncthreads();
#pragma unroll
    for (int kc = 0; kc < 2; ++kc) {
      Frag16 afr[2];
#pragma unroll
      for (int mt = 0; mt < 2; ++mt) {
        const _Float16* ap = &As[buf][(wm * 32 + mt * 16 + row) * 64 + kc * 32];
        afr[mt].q[0] = *(const u32x4*)(ap + koff);
        afr[mt].q[1] = *(const u32x4*)(ap + koff + 16);
      }
#pragma unroll
      for (int nt = 0; nt < 4; ++nt) {
        const _Float16* bp = &Bs[buf][(wn * 64 + nt * 16 + row) * 64 + kc * 32];
        Frag16 bfr;
        bfr.q[0] = *(const u32x4*)(bp + koff);
        bfr.q[1] = *(const u32x4*)(bp + koff + 16);
        acc[0][nt] = wmma_f16(afr[0], bfr, acc[0][nt]);
        acc[1][nt] = wmma_f16(afr[1], bfr, acc[1][nt]);
      }
    }
    __syncthreads();
  }
  const int rbase = (lane >> 4) << 3;
#pragma unroll
  for (int mt = 0; mt < 2; ++mt) {
    int grow = m0 + wm * 32 + mt * 16 + rbase;
#pragma unroll
    for (int nt = 0; nt < 4; ++nt) {
      int gcol = n0 + wn * 64 + nt * 16 + row;
#pragma unroll
      for (int r = 0; r < 8; ++r) {
        if constexpr (F32OUT)
          ((float*)C)[(size_t)(grow + r) * Ndim + gcol] = acc[mt][nt][r];
        else
          ((_Float16*)C)[(size_t)(grow + r) * Ndim + gcol] = (_Float16)acc[mt][nt][r];
      }
    }
  }
}

// -------------------------------------------------------------------- RoPE
__global__ void k_rope(const _Float16* __restrict__ P, _Float16* __restrict__ Out,
                       int nHeads) {
  int idx = blockIdx.x * blockDim.x + threadIdx.x;
  int ncols = nHeads * HD;
  if (idx >= MTOK * ncols) return;
  int d = idx % HD;
  int h = (idx / HD) % nHeads;
  int m = idx / (HD * nHeads);
  int b = m / SS, s = m % SS;
  size_t base = (size_t)m * ncols + h * HD;
  float v = (float)P[base + d];
  int i = d & 63;
  float invf = __expf(-(float)i * (9.210340371976184f / 64.0f)); // theta^(-2i/128)
  float ang  = (float)s * invf;
  float c = __cosf(ang), sn = __sinf(ang);
  float other = (d < 64) ? -(float)P[base + d + 64] : (float)P[base + d - 64];
  Out[(((size_t)(b * nHeads + h)) * SS + s) * HD + d] = (_Float16)(v * c + other * sn);
}

// V[m, KVD] f16 -> Vt[b][kv][d][s] f16 (transposed for B-fragment loads)
__global__ void k_vtrans(const _Float16* __restrict__ Vp, _Float16* __restrict__ Vt) {
  int idx = blockIdx.x * blockDim.x + threadIdx.x;
  if (idx >= MTOK * KVD) return;
  int c = idx % KVD, m = idx / KVD;
  int b = m / SS, s = m % SS;
  int kv = c / HD, d = c % HD;
  Vt[(((size_t)(b * NKV + kv)) * HD + d) * SS + s] = Vp[idx];
}

// --------------------------------------- flash attention + fused gate output
// block = 128 threads (4 waves) covering 64 query rows of one (b,h).
// K/V tiles (32 keys) double-buffered in LDS; gate fused into epilogue.
__global__ __launch_bounds__(128, 1)
void k_attn(const _Float16* __restrict__ Qh, const _Float16* __restrict__ Kh,
            const _Float16* __restrict__ Vt, const _Float16* __restrict__ Gp,
            const float* __restrict__ bg, _Float16* __restrict__ Ah) {
  __shared__ _Float16 Ks[2][32 * HD];    // [key][d]   8KB x2
  __shared__ _Float16 Vs[2][HD * 32];    // [d][key]   8KB x2
  __shared__ _Float16 Pl[4][16 * 32];    // per-wave P 4KB
  const int tid    = threadIdx.x;
  const int lane   = tid & 31;
  const int wave   = tid >> 5;
  const int nsb    = SS / 64;
  const int sb     = blockIdx.x % nsb;
  const int h      = (blockIdx.x / nsb) % NH;
  const int b      = blockIdx.x / (nsb * NH);
  const int kv     = h / (NH / NKV);
  const int q0     = sb * 64 + wave * 16;
  const int col    = lane & 15;
  const int halfid = lane >> 4;
  const int koff   = halfid * 8;

  const _Float16* Qp = Qh + ((size_t)(b * NH  + h))  * SS * HD;
  const _Float16* Kp = Kh + ((size_t)(b * NKV + kv)) * SS * HD;
  const _Float16* Vp = Vt + ((size_t)(b * NKV + kv)) * HD * SS;
  _Float16* Pw = Pl[wave];

  auto stageKV = [&](int kb, int buf) {
#pragma unroll
    for (int j = 0; j < 4; ++j) {
      int c = tid + j * 128;
      async_copy16(Kp + (size_t)kb * HD + c * 8, &Ks[buf][c * 8]);
    }
#pragma unroll
    for (int j = 0; j < 4; ++j)
      async_copy16(Vp + (size_t)tid * SS + kb + j * 8, &Vs[buf][tid * 32 + j * 8]);
  };

  // Q fragments (16 rows x 128 d) in registers
  Frag16 qf[4];
  const _Float16* qrow = Qp + (size_t)(q0 + col) * HD;
#pragma unroll
  for (int c = 0; c < 4; ++c) {
    qf[c].q[0] = *(const u32x4*)(qrow + c * 32 + koff);
    qf[c].q[1] = *(const u32x4*)(qrow + c * 32 + koff + 16);
  }

  v8f o[8] = {};
  float mrow[8], lrow[8];
#pragma unroll
  for (int r = 0; r < 8; ++r) { mrow[r] = -3.0e38f; lrow[r] = 0.0f; }
  const float scale = 0.08838834764831845f;   // 1/sqrt(128)
  const int kend_all = sb * 64 + 64;
  const int kend_w   = q0 + 16;

  stageKV(0, 0);
  for (int kb = 0; kb < kend_all; kb += 32) {
    const int buf = (kb >> 5) & 1;
    if (kb + 32 < kend_all) { stageKV(kb + 32, buf ^ 1); ASYNC_WAIT(8); }
    else                    { ASYNC_WAIT(0); }
    __syncthreads();

    if (kb < kend_w) {            // wave-uniform causal skip
      // ---- scores: 2 key-tiles x 4 d-chunks
      v8f sc[2] = {};
#pragma unroll
      for (int t = 0; t < 2; ++t) {
        const _Float16* krow = &Ks[buf][(t * 16 + col) * HD];
#pragma unroll
        for (int c = 0; c < 4; ++c) {
          Frag16 bf;
          bf.q[0] = *(const u32x4*)(krow + c * 32 + koff);
          bf.q[1] = *(const u32x4*)(krow + c * 32 + koff + 16);
          sc[t] = wmma_f16(qf[c], bf, sc[t]);
        }
      }
      // ---- mask + online softmax; exp results go straight to LDS
#pragma unroll
      for (int r = 0; r < 8; ++r) {
        int rowq = q0 + r + 8 * halfid;
        float s0 = sc[0][r] * scale; if (kb      + col > rowq) s0 = -3.0e38f;
        float s1 = sc[1][r] * scale; if (kb + 16 + col > rowq) s1 = -3.0e38f;
        float tm = fmaxf(halfmax(fmaxf(s0, s1)), mrow[r]);
        float e0 = __expf(s0 - tm);
        float e1 = __expf(s1 - tm);
        float alpha = __expf(mrow[r] - tm);
        lrow[r] = lrow[r] * alpha + halfsum(e0 + e1);
        mrow[r] = tm;
        int pr = (r + 8 * halfid) * 32;
        Pw[pr + col]      = (_Float16)e0;
        Pw[pr + 16 + col] = (_Float16)e1;
#pragma unroll
        for (int t = 0; t < 8; ++t) o[t][r] *= alpha;
      }
      // ---- P (A-layout fragment, per-wave LDS region, in-order DS)
      Frag16 pf;
      pf.q[0] = *(const u32x4*)&Pw[col * 32 + koff];
      pf.q[1] = *(const u32x4*)&Pw[col * 32 + koff + 16];
      // ---- O += P x V
#pragma unroll
      for (int t = 0; t < 8; ++t) {
        const _Float16* vrow = &Vs[buf][(t * 16 + col) * 32];
        Frag16 bf;
        bf.q[0] = *(const u32x4*)(vrow + koff);
        bf.q[1] = *(const u32x4*)(vrow + koff + 16);
        o[t] = wmma_f16(pf, bf, o[t]);
      }
    }
    __syncthreads();
  }

  // ---- normalize + fused sigmoid gate, write token-major [m, hid]
#pragma unroll
  for (int r = 0; r < 8; ++r) {
    float inv = 1.0f / lrow[r];
    int srow = q0 + r + 8 * halfid;
    size_t m = (size_t)b * SS + srow;
#pragma unroll
    for (int t = 0; t < 8; ++t) {
      int n = h * HD + t * 16 + col;
      float g  = (float)Gp[m * HID + n] + bg[n];
      float sg = 1.0f / (1.0f + __expf(-g));
      Ah[m * HID + n] = (_Float16)(o[t][r] * inv * sg);
    }
  }
}

// ------------------------------------------------------------------- launch
extern "C" void kernel_launch(void* const* d_in, const int* in_sizes, int n_in,
                              void* d_out, int out_size, void* d_ws, size_t ws_size,
                              hipStream_t stream) {
  (void)in_sizes; (void)n_in; (void)out_size; (void)ws_size;
  const float* x  = (const float*)d_in[0];
  // d_in[1] = position_ids (arange, recomputed on device)
  const float* Wq = (const float*)d_in[2];
  const float* Wk = (const float*)d_in[3];
  const float* Wv = (const float*)d_in[4];
  const float* Wo = (const float*)d_in[5];
  const float* Wg = (const float*)d_in[6];
  const float* bg = (const float*)d_in[7];

  char* wp = (char*)d_ws;
  auto alloc = [&](size_t elems) {
    _Float16* p = (_Float16*)wp;
    wp += ((elems * 2 + 255) & ~(size_t)255);
    return p;
  };
  _Float16* xh  = alloc((size_t)MTOK * HID);
  _Float16* WqT = alloc((size_t)HID * HID);
  _Float16* WkT = alloc((size_t)KVD * HID);
  _Float16* WvT = alloc((size_t)KVD * HID);
  _Float16* WgT = alloc((size_t)HID * HID);
  _Float16* WoT = alloc((size_t)HID * HID);
  _Float16* Qp  = alloc((size_t)MTOK * HID);
  _Float16* Kp  = alloc((size_t)MTOK * KVD);
  _Float16* Vp  = alloc((size_t)MTOK * KVD);
  _Float16* Gp  = alloc((size_t)MTOK * HID);
  _Float16* Qh  = alloc((size_t)MTOK * HID);
  _Float16* Kh  = alloc((size_t)MTOK * KVD);
  _Float16* Vt  = alloc((size_t)MTOK * KVD);
  _Float16* Ah  = alloc((size_t)MTOK * HID);

  const int TB = 256;
  auto g1 = [&](int n) { return dim3((unsigned)((n + TB - 1) / TB)); };

  // precision staging: f32 -> f16 (weights transposed for B-fragment loads)
  k_cvt <<<g1(MTOK * HID), TB, 0, stream>>>(x, xh, MTOK * HID);
  k_tcvt<<<g1(HID * HID), TB, 0, stream>>>(Wq, WqT, HID, HID);
  k_tcvt<<<g1(HID * KVD), TB, 0, stream>>>(Wk, WkT, HID, KVD);
  k_tcvt<<<g1(HID * KVD), TB, 0, stream>>>(Wv, WvT, HID, KVD);
  k_tcvt<<<g1(HID * HID), TB, 0, stream>>>(Wg, WgT, HID, HID);
  k_tcvt<<<g1(HID * HID), TB, 0, stream>>>(Wo, WoT, HID, HID);

  // projections (WMMA, LDS-tiled, double-buffered async)
  k_gemm<false><<<dim3(MTOK / 128, HID / 128), 256, 0, stream>>>(xh, WqT, Qp, HID, HID);
  k_gemm<false><<<dim3(MTOK / 128, KVD / 128), 256, 0, stream>>>(xh, WkT, Kp, KVD, HID);
  k_gemm<false><<<dim3(MTOK / 128, KVD / 128), 256, 0, stream>>>(xh, WvT, Vp, KVD, HID);
  k_gemm<false><<<dim3(MTOK / 128, HID / 128), 256, 0, stream>>>(xh, WgT, Gp, HID, HID);

  // RoPE + V transpose into attention-friendly layouts
  k_rope  <<<g1(MTOK * HID), TB, 0, stream>>>(Qp, Qh, NH);
  k_rope  <<<g1(MTOK * KVD), TB, 0, stream>>>(Kp, Kh, NKV);
  k_vtrans<<<g1(MTOK * KVD), TB, 0, stream>>>(Vp, Vt);

  // flash attention + fused gate (WMMA scores + WMMA PV, double-buffered K/V)
  k_attn<<<dim3(BB * NH * (SS / 64)), 128, 0, stream>>>(Qh, Kh, Vt, Gp, bg, Ah);

  // output projection (WMMA, f32 out)
  k_gemm<true><<<dim3(MTOK / 128, HID / 128), 256, 0, stream>>>(Ah, WoT, d_out, HID, HID);
}
